// Aggregator1_26886495273086
// MI455X (gfx1250) — matrix operand order repaired and checked by reading, not compiled, not run on hardware
//
#include <hip/hip_runtime.h>

typedef __bf16 bf16;
typedef __attribute__((ext_vector_type(16))) __bf16 v16bf;
typedef __attribute__((ext_vector_type(8)))  __bf16 v8bf;
typedef __attribute__((ext_vector_type(8)))  float  v8f;

#define DDIM 128

// ---------------------------------------------------------------------------
// fp32 -> bf16 elementwise conversion
// ---------------------------------------------------------------------------
__global__ void cvt_f32_bf16(const float* __restrict__ in, bf16* __restrict__ out, long n) {
    long i = (long)blockIdx.x * blockDim.x + threadIdx.x;
    if (i < n) out[i] = (bf16)in[i];
}

// ---------------------------------------------------------------------------
// W1eff[n,k] = sum_j w1[n, 128+j] * wa_v[j, k]   (fold 2nd half of w1 with wa_v)
// grid = 128 (n), block = 128 (k)
// ---------------------------------------------------------------------------
__global__ void fold_w1(const float* __restrict__ w1, const float* __restrict__ wav,
                        float* __restrict__ weff) {
    int n = blockIdx.x, k = threadIdx.x;
    float acc = 0.f;
    for (int j = 0; j < DDIM; ++j)
        acc += w1[n * 2 * DDIM + DDIM + j] * wav[j * DDIM + k];
    weff[n * DDIM + k] = acc;
}

// ---------------------------------------------------------------------------
// Pack a 128x128 weight into WMMA B-fragment order (bf16).
// Fragment layout (16-bit B, 32x16 chunk): lane L holds column n = nblk*16 +
// (L&15); K half per lane group: khalf = (L<16)?0:16; element e -> K =
// kc*32 + khalf + e.  Flat order: (((nblk*4 + kc)*32 + lane)*16 + e).
// B[k,n] = transposed ? src[n*ld + k] : src[k*ld + n]
// grid = 64, block = 256  (16384 elements)
// ---------------------------------------------------------------------------
__global__ void pack_b(const float* __restrict__ src, int ld, int transposed,
                       bf16* __restrict__ dst) {
    int t    = blockIdx.x * blockDim.x + threadIdx.x;
    int e    = t & 15;
    int lane = (t >> 4) & 31;
    int kc   = (t >> 9) & 3;
    int nblk = (t >> 11) & 7;
    int n = nblk * 16 + (lane & 15);
    int k = kc * 32 + ((lane >> 4) << 4) + e;
    float v = transposed ? src[n * ld + k] : src[k * ld + n];
    dst[t] = (bf16)v;
}

// ---------------------------------------------------------------------------
// CSR segment sum of elementwise products over bf16 operands, fp32 accumulate.
// One block of 64 threads per node; each lane owns a bf16 feature PAIR so the
// row gather is one b32 per lane (64 lanes -> 256B coalesced request).
// out[i, :] = bf16( sum_{e in [ptr[i], ptr[i+1])} X[la[e], :] * Y[lb[e], :] )
// Next-edge rows prefetched (global_prefetch_b8) -- this is the HBM roofline
// phase, and bf16 operands halve its traffic vs fp32.
// ---------------------------------------------------------------------------
__global__ void seg_mul_sum(const bf16* __restrict__ X, const bf16* __restrict__ Y,
                            const int* __restrict__ ptr, const int* __restrict__ la,
                            const int* __restrict__ lb, bf16* __restrict__ out) {
    int i  = blockIdx.x;
    int j2 = threadIdx.x * 2;            // feature pair
    int s = ptr[i], t = ptr[i + 1];
    float accx = 0.f, accy = 0.f;
    for (int e = s; e < t; ++e) {
        if (e + 1 < t) {
            int an = la[e + 1], bn = lb[e + 1];
            __builtin_prefetch(&X[(long)an * DDIM + j2], 0, 0);
            __builtin_prefetch(&Y[(long)bn * DDIM + j2], 0, 0);
        }
        int a = la[e], b = lb[e];
        unsigned wx = *(const unsigned*)(X + (long)a * DDIM + j2);
        unsigned wy = *(const unsigned*)(Y + (long)b * DDIM + j2);
        // exact bf16 -> fp32: bits << 16
        float xl = __uint_as_float(wx << 16);
        float xh = __uint_as_float(wx & 0xffff0000u);
        float yl = __uint_as_float(wy << 16);
        float yh = __uint_as_float(wy & 0xffff0000u);
        accx += xl * yl;
        accy += xh * yh;
    }
    bf16 o0 = (bf16)accx, o1 = (bf16)accy;       // RNE converts
    unsigned u0 = (unsigned)__builtin_bit_cast(unsigned short, o0);
    unsigned u1 = (unsigned)__builtin_bit_cast(unsigned short, o1);
    *(unsigned*)(out + (long)i * DDIM + j2) = (u1 << 16) | u0;
}

// ---------------------------------------------------------------------------
// WMMA GEMM:  C[M x 128] = A1 @ B1 (+ A2 @ B2), A row-major bf16 (K=128),
// B pre-packed bf16 fragments staged in LDS, output fp32 (C) and/or bf16 (Cbf).
// Block = 256 threads = 8 waves; each wave owns a 16-row x 128-col strip
// (8 accumulator tiles), K in 4 chunks of 32 -> 32/64 v_wmma per wave.
// ---------------------------------------------------------------------------
__global__ __launch_bounds__(256)
void gemm_bf16(const bf16* __restrict__ A1, const bf16* __restrict__ B1,
               const bf16* __restrict__ A2, const bf16* __restrict__ B2,
               float* __restrict__ C, bf16* __restrict__ Cbf, int M) {
    extern __shared__ char smem_raw[];
    bf16* sB = (bf16*)smem_raw;            // [16384] B1, optionally [16384] B2

    // cooperative stage: 32KB per matrix = 2048 x 16B
    {
        uint4*       sdst = (uint4*)smem_raw;
        const uint4* g1   = (const uint4*)B1;
        for (int i = threadIdx.x; i < 2048; i += 256) sdst[i] = g1[i];
        if (A2 != nullptr) {
            const uint4* g2 = (const uint4*)B2;
            for (int i = threadIdx.x; i < 2048; i += 256) sdst[2048 + i] = g2[i];
        }
    }
    __syncthreads();   // uniform: before any wave can exit

    const int lane = threadIdx.x & 31;
    const int wave = threadIdx.x >> 5;
    const int row0 = blockIdx.x * 128 + wave * 16;
    if (row0 >= M) return;  // uniform per wave: EXEC stays all-ones inside

    const int r  = lane & 15;          // A row within tile
    const int kb = (lane >> 4) * 8;    // A K-offset within chunk is 2*kb (0 or 16)
    int arow = row0 + r;
    if (arow >= M) arow = M - 1;       // clamp loads; stores guarded below

    const bf16* a1p = A1 + (long)arow * DDIM;

    v8f acc[8] = {};

    #pragma unroll
    for (int kc = 0; kc < 4; ++kc) {
        // A fragment: elements 0..15 = K kc*32+2*kb.. (contiguous)
        v8bf lo = *(const v8bf*)(a1p + kc * 32 + 2 * kb);
        v8bf hi = *(const v8bf*)(a1p + kc * 32 + 2 * kb + 8);
        v16bf af = __builtin_shufflevector(lo, hi, 0,1,2,3,4,5,6,7,8,9,10,11,12,13,14,15);
        #pragma unroll
        for (int nb = 0; nb < 8; ++nb) {
            v16bf bf_ = *(const v16bf*)(sB + (((nb * 4 + kc) * 32 + lane) * 16));
            acc[nb] = __builtin_amdgcn_wmma_f32_16x16x32_bf16(
                false, af, false, bf_, (short)0, acc[nb], false, false);
        }
    }

    if (A2 != nullptr) {
        const bf16* a2p = A2 + (long)arow * DDIM;
        const bf16* sB2 = sB + 16384;
        #pragma unroll
        for (int kc = 0; kc < 4; ++kc) {
            v8bf lo = *(const v8bf*)(a2p + kc * 32 + 2 * kb);
            v8bf hi = *(const v8bf*)(a2p + kc * 32 + 2 * kb + 8);
            v16bf af = __builtin_shufflevector(lo, hi, 0,1,2,3,4,5,6,7,8,9,10,11,12,13,14,15);
            #pragma unroll
            for (int nb = 0; nb < 8; ++nb) {
                v16bf bf_ = *(const v16bf*)(sB2 + (((nb * 4 + kc) * 32 + lane) * 16));
                acc[nb] = __builtin_amdgcn_wmma_f32_16x16x32_bf16(
                    false, af, false, bf_, (short)0, acc[nb], false, false);
            }
        }
    }

    // C/D layout: VGPR i -> row M = i + (lane<16 ? 0 : 8), col N = lane&15
    const int ccol = lane & 15;
    const int crow = (lane >> 4) * 8;
    if (C != nullptr) {
        #pragma unroll
        for (int nb = 0; nb < 8; ++nb) {
            #pragma unroll
            for (int i = 0; i < 8; ++i) {
                int m = row0 + crow + i;
                if (m < M) C[(long)m * DDIM + nb * 16 + ccol] = acc[nb][i];
            }
        }
    }
    if (Cbf != nullptr) {
        #pragma unroll
        for (int nb = 0; nb < 8; ++nb) {
            #pragma unroll
            for (int i = 0; i < 8; ++i) {
                int m = row0 + crow + i;
                if (m < M) Cbf[(long)m * DDIM + nb * 16 + ccol] = (bf16)acc[nb][i];
            }
        }
    }
}

// ---------------------------------------------------------------------------
extern "C" void kernel_launch(void* const* d_in, const int* in_sizes, int n_in,
                              void* d_out, int out_size, void* d_ws, size_t ws_size,
                              hipStream_t stream) {
    const float* t_embed  = (const float*)d_in[0];
    const float* v_embed  = (const float*)d_in[1];
    const float* a_embed  = (const float*)d_in[2];
    const float* wt       = (const float*)d_in[3];
    const float* wa_t     = (const float*)d_in[4];
    const float* wa_v     = (const float*)d_in[5];
    const float* w1       = (const float*)d_in[6];
    const float* w2       = (const float*)d_in[7];
    const float* wa       = (const float*)d_in[8];
    const int*   ptr_t    = (const int*)d_in[9];
    const int*   a_list_t = (const int*)d_in[10];
    const int*   v_list_t = (const int*)d_in[11];
    const int*   ptr_v    = (const int*)d_in[12];
    const int*   a_list_v = (const int*)d_in[13];
    const int*   t_list_v = (const int*)d_in[14];

    const int NT = in_sizes[0] / DDIM;
    const int NV = in_sizes[1] / DDIM;
    const int NA = in_sizes[2] / DDIM;

    // workspace carve-up (256B aligned slices)
    char* wsp = (char*)d_ws;
    auto alloc = [&](size_t bytes) -> char* {
        char* p = wsp;
        wsp += (bytes + 255) & ~(size_t)255;
        return p;
    };
    float* W1eff = (float*)alloc((size_t)DDIM * DDIM * 4);
    bf16*  Bfr[7];
    for (int i = 0; i < 7; ++i) Bfr[i] = (bf16*)alloc((size_t)DDIM * DDIM * 2);
    bf16*  t_bf   = (bf16*)alloc((size_t)NT * DDIM * 2);
    bf16*  v_bf   = (bf16*)alloc((size_t)NV * DDIM * 2);
    bf16*  a_bf   = (bf16*)alloc((size_t)NA * DDIM * 2);
    bf16*  Tp_bf  = (bf16*)alloc((size_t)NT * DDIM * 2);  // bf16(t_embed @ wt.T)
    bf16*  Ap_bf  = (bf16*)alloc((size_t)NA * DDIM * 2);  // bf16(a_embed @ wa_t.T)
    bf16*  pre_t  = (bf16*)alloc((size_t)NT * DDIM * 2);  // tuple-side segment sum
    bf16*  pre_v  = (bf16*)alloc((size_t)NV * DDIM * 2);  // value-side segment sum

    float* out_t = (float*)d_out;
    float* out_v = out_t + (size_t)NT * DDIM;
    float* out_a = out_v + (size_t)NV * DDIM;

    const size_t LDS1 = 32768;   // single-source GEMM: B1 only
    const size_t LDS2 = 65536;   // dual-source GEMM: B1 + B2

    // 1. fold w1[:,128:] @ wa_v
    fold_w1<<<DDIM, DDIM, 0, stream>>>(w1, wa_v, W1eff);

    // 2. pack all B operands into WMMA fragment order
    pack_b<<<64, 256, 0, stream>>>(wt,        DDIM,     1, Bfr[0]); // T' = t @ wt.T
    pack_b<<<64, 256, 0, stream>>>(wa_t,      DDIM,     1, Bfr[1]); // A' = a @ wa_t.T
    pack_b<<<64, 256, 0, stream>>>(wa,        DDIM,     0, Bfr[2]); // a_out = a @ wa
    pack_b<<<64, 256, 0, stream>>>(w1,        2 * DDIM, 1, Bfr[3]); // t part of w1
    pack_b<<<64, 256, 0, stream>>>(W1eff,     DDIM,     1, Bfr[4]); // folded seg part
    pack_b<<<64, 256, 0, stream>>>(w2,        2 * DDIM, 1, Bfr[5]); // v part of w2
    pack_b<<<64, 256, 0, stream>>>(w2 + DDIM, 2 * DDIM, 1, Bfr[6]); // seg part of w2

    // 3. bf16 copies of embeddings (WMMA A operands + bf16 gather sources)
    {
        long n;
        n = (long)NT * DDIM; cvt_f32_bf16<<<(int)((n + 255) / 256), 256, 0, stream>>>(t_embed, t_bf, n);
        n = (long)NV * DDIM; cvt_f32_bf16<<<(int)((n + 255) / 256), 256, 0, stream>>>(v_embed, v_bf, n);
        n = (long)NA * DDIM; cvt_f32_bf16<<<(int)((n + 255) / 256), 256, 0, stream>>>(a_embed, a_bf, n);
    }

    // 4. hoisted per-node linear maps (bf16 out: only the gather phase reads them)
    gemm_bf16<<<(NT + 127) / 128, 256, LDS1, stream>>>(t_bf, Bfr[0], nullptr, nullptr,
                                                       nullptr, Tp_bf, NT);
    gemm_bf16<<<(NA + 127) / 128, 256, LDS1, stream>>>(a_bf, Bfr[1], nullptr, nullptr,
                                                       nullptr, Ap_bf, NA);

    // 5. a_out (independent, fp32 final output)
    gemm_bf16<<<(NA + 127) / 128, 256, LDS1, stream>>>(a_bf, Bfr[2], nullptr, nullptr,
                                                       out_a, nullptr, NA);

    // 6. edge gather + CSR segment sums (bf16 operands, fp32 accumulate)
    seg_mul_sum<<<NT, 64, 0, stream>>>(a_bf, v_bf, ptr_t, a_list_t, v_list_t, pre_t);
    seg_mul_sum<<<NV, 64, 0, stream>>>(Ap_bf, Tp_bf, ptr_v, a_list_v, t_list_v, pre_v);

    // 7. fused output GEMMs: X @ W_first.T + pre @ W_second.T  (fp32 final out)
    gemm_bf16<<<(NT + 127) / 128, 256, LDS2, stream>>>(t_bf, Bfr[3], pre_t, Bfr[4],
                                                       out_t, nullptr, NT);
    gemm_bf16<<<(NV + 127) / 128, 256, LDS2, stream>>>(v_bf, Bfr[5], pre_v, Bfr[6],
                                                       out_v, nullptr, NV);
}
